// SelfAttention_87651692577010
// MI455X (gfx1250) — matrix operand order repaired and checked by reading, not compiled, not run on hardware
//
#include <hip/hip_runtime.h>

// ---------------------------------------------------------------------------
// Windowed self-attention (all 8 heads attend over contiguous 512-token
// windows) for MI455X / gfx1250, wave32 + WMMA f16.
//   stage 0: w_qkv, w_proj  -> transposed f16 (Bt layout, N x K)
//   stage 1: qkv = x @ w_qkv                 (f32 A staged->f16, f16 out)
//   stage 2: flash attention, transposed (S^T = K Q^T, O^T = V^T P^T) so the
//            softmax reduction is over VGPRs (1 shfl per query tile)
//   stage 3: out = attn @ w_proj + b_proj    (f16 A, f32 out)
// ---------------------------------------------------------------------------

typedef __attribute__((ext_vector_type(16))) _Float16 v16h;
typedef __attribute__((ext_vector_type(8)))  float    v8f;

union Frag16 { v16h v; uint4 q[2]; };
union Frag8  { uint4 q; _Float16 h[8]; };
union Half4  { uint2 u2; _Float16 h[4]; };

__device__ __forceinline__ v8f wmma_f16(v16h a, v16h b, v8f c) {
  // D(16x16 f32) = A(16x32 f16) * B(32x16 f16) + C
  return __builtin_amdgcn_wmma_f32_16x16x32_f16(false, a, false, b,
                                                (short)0, c, false, false);
}

// ---------------------------------------------------------------------------
// Prep: dst[n*K + k] = (f16) src[k*N + n]   (transpose + convert)
// ---------------------------------------------------------------------------
__global__ __launch_bounds__(256) void transpose_cvt(
    const float* __restrict__ src, _Float16* __restrict__ dst, int K, int N) {
  int idx = blockIdx.x * 256 + threadIdx.x;
  if (idx < K * N) {
    int n = idx / K, k = idx - n * K;
    dst[idx] = (_Float16)src[(size_t)k * N + n];
  }
}

// ---------------------------------------------------------------------------
// GEMM: Out[M x Ntot] = A[M x Ktot] * Bt[Ntot x Ktot]^T (+bias)
// Block tile 128x64, 8 waves (4x2), wave tile 32x32 (2x2 WMMA tiles), BK=32.
// ---------------------------------------------------------------------------
template <bool A_F32, bool OUT_F32>
__global__ __launch_bounds__(256) void gemm128x64(
    const void* __restrict__ Ap, const _Float16* __restrict__ Bt,
    void* __restrict__ Outp, const float* __restrict__ bias,
    int Ktot, int Ntot) {
  constexpr int LDS_BYTES = OUT_F32 ? (128 * 72 * 4) : (128 * 72 * 2);
  __shared__ __align__(16) char smem[LDS_BYTES];
  _Float16* As = (_Float16*)smem;     // [128][40] padded (bank-friendly)
  _Float16* Bs = As + 128 * 40;       // [64][40]

  const int tid = threadIdx.x;
  const int lane = tid & 31, wid = tid >> 5;
  const int q = lane & 15, hi = lane >> 4;
  const int wm = wid >> 1, wn = wid & 1;           // 4 x 2 wave grid
  const int m0 = blockIdx.x * 128, n0 = blockIdx.y * 64;

  v8f acc[2][2] = {};

  for (int kk = 0; kk < Ktot; kk += 32) {
    __syncthreads();
    if (A_F32) {
      const float* A = (const float*)Ap;
      #pragma unroll
      for (int i = 0; i < 4; ++i) {                 // 128 rows x 32 f32
        int ci = i * 256 + tid;
        int r = ci >> 3, c = (ci & 7) * 4;
        const float* gp = A + (size_t)(m0 + r) * Ktot + kk + c;
        float4 f = *(const float4*)gp;
        if (kk + 32 < Ktot) __builtin_prefetch(gp + 32, 0, 1);
        Half4 pk;
        pk.h[0] = (_Float16)f.x; pk.h[1] = (_Float16)f.y;
        pk.h[2] = (_Float16)f.z; pk.h[3] = (_Float16)f.w;
        *(uint2*)(As + r * 40 + c) = pk.u2;
      }
    } else {
      const _Float16* A = (const _Float16*)Ap;
      #pragma unroll
      for (int i = 0; i < 2; ++i) {                 // 128 rows x 32 f16
        int ci = i * 256 + tid;
        int r = ci >> 2, c = (ci & 3) * 8;
        const _Float16* gp = A + (size_t)(m0 + r) * Ktot + kk + c;
        if (kk + 32 < Ktot) __builtin_prefetch(gp + 32, 0, 1);
        *(uint4*)(As + r * 40 + c) = *(const uint4*)gp;
      }
    }
    {                                               // 64 rows x 32 f16 of Bt
      int r = tid >> 2, c = (tid & 3) * 8;
      *(uint4*)(Bs + r * 40 + c) =
          *(const uint4*)(Bt + (size_t)(n0 + r) * Ktot + kk + c);
    }
    __syncthreads();

    Frag16 a[2], b[2];
    #pragma unroll
    for (int mt = 0; mt < 2; ++mt) {   // A frag: K(e) = (e<8? hi*8+e : 16+hi*8+e-8)
      _Float16* base = As + (wm * 32 + mt * 16 + q) * 40;
      a[mt].q[0] = *(uint4*)(base + hi * 8);
      a[mt].q[1] = *(uint4*)(base + 16 + hi * 8);
    }
    #pragma unroll
    for (int nt = 0; nt < 2; ++nt) {   // B frag: lane=N, K(e) = (e<8? hi*8+e : 16+hi*8+e-8)
      _Float16* base = Bs + (wn * 32 + nt * 16 + q) * 40;
      b[nt].q[0] = *(uint4*)(base + hi * 8);
      b[nt].q[1] = *(uint4*)(base + 16 + hi * 8);
    }
    #pragma unroll
    for (int mt = 0; mt < 2; ++mt)
      #pragma unroll
      for (int nt = 0; nt < 2; ++nt)
        acc[mt][nt] = wmma_f16(a[mt].v, b[nt].v, acc[mt][nt]);
  }

  __syncthreads();  // done reading stage buffers; reuse smem for the epilogue
  if (OUT_F32) {
    float* OutS = (float*)smem;  // [128][72]
    #pragma unroll
    for (int mt = 0; mt < 2; ++mt)
      #pragma unroll
      for (int nt = 0; nt < 2; ++nt)
        #pragma unroll
        for (int r = 0; r < 8; ++r)
          OutS[(wm * 32 + mt * 16 + r + 8 * hi) * 72 + wn * 32 + nt * 16 + q] =
              acc[mt][nt][r];
    __syncthreads();
    float* O = (float*)Outp;
    #pragma unroll
    for (int i = 0; i < 8; ++i) {
      int ci = i * 256 + tid;
      int r = ci >> 4, c = (ci & 15) * 4;
      float4 v = *(float4*)(OutS + r * 72 + c);
      if (bias) {
        v.x += bias[n0 + c + 0]; v.y += bias[n0 + c + 1];
        v.z += bias[n0 + c + 2]; v.w += bias[n0 + c + 3];
      }
      *(float4*)(O + (size_t)(m0 + r) * Ntot + n0 + c) = v;
    }
  } else {
    _Float16* OutS = (_Float16*)smem;  // [128][72]
    #pragma unroll
    for (int mt = 0; mt < 2; ++mt)
      #pragma unroll
      for (int nt = 0; nt < 2; ++nt)
        #pragma unroll
        for (int r = 0; r < 8; ++r)
          OutS[(wm * 32 + mt * 16 + r + 8 * hi) * 72 + wn * 32 + nt * 16 + q] =
              (_Float16)acc[mt][nt][r];
    __syncthreads();
    _Float16* O = (_Float16*)Outp;
    #pragma unroll
    for (int i = 0; i < 4; ++i) {
      int ci = i * 256 + tid;
      int r = ci >> 3, c = (ci & 7) * 8;
      *(uint4*)(O + (size_t)(m0 + r) * Ntot + n0 + c) =
          *(uint4*)(OutS + r * 72 + c);
    }
  }
}

// ---------------------------------------------------------------------------
// Flash attention over one (b, head, window): 512 tokens, hd = 32.
// 8 waves, each owns 64 query rows.  qkv layout: [token][qsel*256 + h*32 + d].
// Transposed compute: S^T = K Q^T so that in the WMMA C-layout the QUERY is
// the lane index and the KEYS are the VGPR index -> softmax reduction is an
// in-register tree + a single shfl_xor(16).  O^T = V^T P^T keeps the rescale
// factor per-lane.  P^T(f32, C-layout) -> P(f16, B-frag) via a wave-private
// LDS slab using packed b128 stores (keys are contiguous along VGPR index).
// ---------------------------------------------------------------------------
__global__ __launch_bounds__(256) void attn512(
    const _Float16* __restrict__ qkv, _Float16* __restrict__ attn) {
  __shared__ __align__(16) _Float16 Ks[512 * 40];   // K  [key][hd], padded
  __shared__ __align__(16) _Float16 Vs[512 * 40];   // V staging -> P/O slabs
  __shared__ __align__(16) _Float16 Vt[32 * 520];   // V^T [hd][key], padded

  const int tid = threadIdx.x;
  const int lane = tid & 31, wid = tid >> 5;
  const int q = lane & 15, hi = lane >> 4;
  const int blk = blockIdx.x;
  const int b = blk >> 6, rem = blk & 63, h = rem >> 3, win = rem & 7;
  const size_t token0 = (size_t)b * 4096 + (size_t)win * 512;

  const _Float16* Qg = qkv + token0 * 768 + h * 32;
  const _Float16* Kg = Qg + 256;
  const _Float16* Vg = Qg + 512;

  // ---- stage K and V (512 rows x 64B each, coalesced) ----
  #pragma unroll
  for (int i = 0; i < 8; ++i) {
    int ci = i * 256 + tid;
    int r = ci >> 2, c = (ci & 3) * 8;
    *(uint4*)(Ks + r * 40 + c) = *(const uint4*)(Kg + (size_t)r * 768 + c);
    *(uint4*)(Vs + r * 40 + c) = *(const uint4*)(Vg + (size_t)r * 768 + c);
  }
  __syncthreads();
  // ---- transpose V into Vt: 8 strided b16 reads -> one packed b128 write ----
  #pragma unroll
  for (int i = 0; i < 8; ++i) {
    int g = i * 256 + tid;                // 2048 groups of 8 consecutive keys
    int d = g >> 6, kb = (g & 63) * 8;
    Frag8 u;
    #pragma unroll
    for (int e = 0; e < 8; ++e) u.h[e] = Vs[(kb + e) * 40 + d];
    *(uint4*)(Vt + d * 520 + kb) = u.q;
  }
  __syncthreads();  // Vs is now free: becomes per-wave P / O slab

  // ---- Q as B-fragments (lane = query, K-dim = hd), loaded ONCE, pre-scaled
  const int rowb = wid * 64;
  Frag16 qB[4];
  #pragma unroll
  for (int mq = 0; mq < 4; ++mq) {
    const _Float16* base = Qg + (size_t)(rowb + mq * 16 + q) * 768;
    qB[mq].q[0] = *(const uint4*)(base + hi * 8);
    qB[mq].q[1] = *(const uint4*)(base + 16 + hi * 8);
    #pragma unroll
    for (int e = 0; e < 16; ++e)
      qB[mq].v[e] = qB[mq].v[e] * (_Float16)0.1767766953f;  // 1/sqrt(32)
  }

  float mrun[4], lrun[4];
  v8f ot[2][4] = {};                       // O^T accum: [hd tile][query tile]
  #pragma unroll
  for (int mq = 0; mq < 4; ++mq) { mrun[mq] = -1e30f; lrun[mq] = 0.f; }

  _Float16* Pb = Vs + wid * 64 * 40;       // wave-private slab [64][40]

  for (int j = 0; j < 16; ++j) {           // 32 keys per step
    Frag16 kA[2], vA[2];
    #pragma unroll
    for (int kt = 0; kt < 2; ++kt) {       // A frag of K: row=key, K-dim=hd
      _Float16* base = Ks + (j * 32 + kt * 16 + q) * 40;
      kA[kt].q[0] = *(uint4*)(base + hi * 8);
      kA[kt].q[1] = *(uint4*)(base + 16 + hi * 8);
    }
    #pragma unroll
    for (int dt = 0; dt < 2; ++dt) {       // A frag of V^T: row=hd, K-dim=key
      _Float16* base = Vt + (dt * 16 + q) * 520 + j * 32;
      vA[dt].q[0] = *(uint4*)(base + hi * 8);
      vA[dt].q[1] = *(uint4*)(base + 16 + hi * 8);
    }
    #pragma unroll
    for (int mq = 0; mq < 4; ++mq) {
      v8f z = {};
      v8f s0 = wmma_f16(kA[0].v, qB[mq].v, z);   // S^T: rows=keys, cols=query
      v8f s1 = wmma_f16(kA[1].v, qB[mq].v, z);
      // row(=query, per-lane) max over the 32 keys: 16 in-register + 1 shfl
      float mx = fmaxf(s0[0], s1[0]);
      #pragma unroll
      for (int r = 1; r < 8; ++r) mx = fmaxf(mx, fmaxf(s0[r], s1[r]));
      mx = fmaxf(mx, __shfl_xor(mx, 16, 32));
      float mold = mrun[mq];
      float mn = fmaxf(mold, mx);
      float f = __expf(mold - mn);
      mrun[mq] = mn;
      Frag8 p0, p1;
      float rs = 0.f;
      #pragma unroll
      for (int r = 0; r < 8; ++r) {
        float a = __expf(s0[r] - mn);
        float c = __expf(s1[r] - mn);
        rs += a + c;
        p0.h[r] = (_Float16)a;
        p1.h[r] = (_Float16)c;
      }
      rs += __shfl_xor(rs, 16, 32);
      lrun[mq] = lrun[mq] * f + rs;
      #pragma unroll
      for (int dt = 0; dt < 2; ++dt)
        #pragma unroll
        for (int r = 0; r < 8; ++r) ot[dt][mq][r] *= f;  // per-lane rescale
      // P^T -> P[query][key] slab: key = kt*16 + 8*hi + r  (r contiguous)
      _Float16* prow = Pb + (mq * 16 + q) * 40 + hi * 8;
      *(uint4*)(prow)      = p0.q;
      *(uint4*)(prow + 16) = p1.q;
      // read back as B-frag of P^T (lane=query, K-dim=key) and accumulate O^T
      Frag16 pB;
      _Float16* pbase = Pb + (mq * 16 + q) * 40;
      pB.q[0] = *(uint4*)(pbase + hi * 8);
      pB.q[1] = *(uint4*)(pbase + 16 + hi * 8);
      #pragma unroll
      for (int dt = 0; dt < 2; ++dt)
        ot[dt][mq] = wmma_f16(vA[dt].v, pB.v, ot[dt][mq]);
    }
  }

  // ---- normalize (per-lane), transpose O^T->O via packed b128 LDS stores ---
  #pragma unroll
  for (int mq = 0; mq < 4; ++mq) {
    float inv = 1.0f / lrun[mq];
    #pragma unroll
    for (int dt = 0; dt < 2; ++dt) {
      Frag8 u;
      #pragma unroll
      for (int r = 0; r < 8; ++r) u.h[r] = (_Float16)(ot[dt][mq][r] * inv);
      // O[query][hd]: hd = dt*16 + 8*hi + r  (r contiguous)
      *(uint4*)(Pb + (mq * 16 + q) * 40 + dt * 16 + hi * 8) = u.q;
    }
  }
  _Float16* Og = attn + (token0 + rowb) * 256 + h * 32;  // [token][h*32+d]
  #pragma unroll
  for (int i = 0; i < 2; ++i) {
    int rr = i * 32 + lane;
    #pragma unroll
    for (int c4 = 0; c4 < 4; ++c4)
      *(uint4*)(Og + (size_t)rr * 256 + c4 * 8) =
          *(uint4*)(Pb + rr * 40 + c4 * 8);
  }
}

// ---------------------------------------------------------------------------
extern "C" void kernel_launch(void* const* d_in, const int* in_sizes, int n_in,
                              void* d_out, int out_size, void* d_ws,
                              size_t ws_size, hipStream_t stream) {
  (void)in_sizes; (void)n_in; (void)out_size; (void)ws_size;
  const float* x     = (const float*)d_in[0];  // (8, 4096, 256)
  const float* wqkv  = (const float*)d_in[1];  // (256, 768)
  const float* wproj = (const float*)d_in[2];  // (256, 256)
  const float* bproj = (const float*)d_in[3];  // (256,)

  char* ws = (char*)d_ws;
  _Float16* qkv    = (_Float16*)(ws);                                   // 48 MB
  _Float16* attn   = (_Float16*)(ws + (size_t)48 * 1024 * 1024);        // 16 MB
  _Float16* wqkvT  = (_Float16*)(ws + (size_t)64 * 1024 * 1024);        // 384 KB
  _Float16* wprojT = (_Float16*)(ws + (size_t)64 * 1024 * 1024 + 393216);

  // stage 0: transpose/convert weights to f16 Bt layout (N x K)
  transpose_cvt<<<(768 * 256 + 255) / 256, 256, 0, stream>>>(wqkv, wqkvT, 256, 768);
  transpose_cvt<<<(256 * 256 + 255) / 256, 256, 0, stream>>>(wproj, wprojT, 256, 256);

  // stage 1: qkv = x @ w_qkv   (M=32768, K=256, N=768)
  gemm128x64<true, false><<<dim3(256, 12), 256, 0, stream>>>(
      x, wqkvT, qkv, nullptr, 256, 768);

  // stage 2: windowed attention, 8*8*8 = 512 windows of 512 tokens
  attn512<<<512, 256, 0, stream>>>(qkv, attn);

  // stage 3: out = attn @ w_proj + b_proj   (M=32768, K=256, N=256)
  gemm128x64<false, true><<<dim3(256, 4), 256, 0, stream>>>(
      attn, wprojT, d_out, bproj, 256, 256);
}
